// MultiHeadAttention_14912126452247
// MI455X (gfx1250) — compile-verified, matrix operand order checked by reading, not run on hardware
//
#include <hip/hip_runtime.h>

typedef unsigned short u16;
typedef unsigned int   u32;
typedef unsigned long long uptr;
typedef __attribute__((ext_vector_type(16))) __bf16 v16bf;
typedef __attribute__((ext_vector_type(8)))  float  v8f;
typedef __attribute__((ext_vector_type(4)))  int    v4i;

#define D_MODEL 1024
#define NUM_HEADS 16
#define DEPTH 64
#define BATCH 4
#define SEQ 2048
#define LLEN 128
#define LSTART 1920

#define AS1 __attribute__((address_space(1)))
#define AS3 __attribute__((address_space(3)))

// ---- CDNA5 async global->LDS copy (ASYNCcnt path), with sync fallback ----
#if defined(__has_builtin)
#  if __has_builtin(__builtin_amdgcn_global_load_async_to_lds_b128)
#    define HAVE_ASYNC_LDS 1
#  endif
#endif
#ifndef HAVE_ASYNC_LDS
#  define HAVE_ASYNC_LDS 0
#endif

#if defined(__has_builtin)
#  if __has_builtin(__builtin_amdgcn_s_wait_asynccnt)
#    define ASYNC_WAIT0() __builtin_amdgcn_s_wait_asynccnt(0)
#  endif
#endif
#ifndef ASYNC_WAIT0
#  define ASYNC_WAIT0() asm volatile("s_wait_asynccnt 0x0" ::: "memory")
#endif

__device__ __forceinline__ u16 f2bf(float f) {
  u32 u = __float_as_uint(f);
  u32 r = u + 0x7FFFu + ((u >> 16) & 1u);   // round-to-nearest-even
  return (u16)(r >> 16);
}

union FragU { v16bf v; uint4 q[2]; };

#define LDSTRIDE 40   // elements (80B): 16B-aligned rows, bank-group friendly

// Copy a ROWS x 32 bf16 tile (row stride ld) into LDS (row stride LDSTRIDE).
// 16B chunks; 4 chunks per row; 256 threads; ROWS compile-time so the loop
// fully unrolls into straight-line async-load issues (no exec-mask loop).
template<int ROWS>
__device__ __forceinline__ void copy_tile(const u16* __restrict__ g, int ld,
                                          u16* lds, int t) {
  constexpr int NCHUNKS = ROWS * 4;          // 512 (A) / 256 or 512 (B)
  #pragma unroll
  for (int c0 = 0; c0 < NCHUNKS; c0 += 256) {
    int c = c0 + t;
    int r = c >> 2, s = c & 3;
    const u16* src = g + (long)r * ld + s * 8;
    u16* dst = lds + r * LDSTRIDE + s * 8;
#if HAVE_ASYNC_LDS
    __builtin_amdgcn_global_load_async_to_lds_b128(
        (AS1 v4i*)(uptr)src, (AS3 v4i*)(u32)(uptr)dst, 0, 0);
#else
    *(uint4*)dst = *(const uint4*)src;
#endif
  }
}

// Load a 16x32 bf16 fragment from an LDS tile (A-layout per ISA 7.12.2):
// lane L: m = L&15, half = L>>4; VGPR0-3 <- K = 8*half.., VGPR4-7 <- K = 16+8*half..
__device__ __forceinline__ v16bf lds_frag(const u16* base, int lane) {
  int m    = lane & 15;
  int half = lane >> 4;
  const u16* p = base + m * LDSTRIDE + half * 8;
  FragU f;
  f.q[0] = *(const uint4*)(p);
  f.q[1] = *(const uint4*)(p + 16);
  return f.v;
}

// Batched GEMM: C[M,N](+bias) = A[M,K](bf16) * BT[N,K](bf16)^T
// Block = 256 threads = 8 waves (4 M x 2 N). Wave tile 32 x (WN*16).
// Block tile 128 x (32*WN). Double-buffered LDS staging of A/B K-slabs (BK=32),
// async copies overlapped with WMMA of the current slab.
template<int WN>
__global__ __launch_bounds__(256) void wmma_gemm_lds(
    const u16* __restrict__ A, const u16* __restrict__ BT,
    float* __restrict__ Cf, u16* __restrict__ Cb,
    const float* __restrict__ bias,
    int K, int lda, int ldb, int ldc,
    long aOffB, long aOffH, long bOffB, long bOffH, long cOffB, long cOffH,
    int nH)
{
  __shared__ alignas(16) u16 As[2][128 * LDSTRIDE];
  __shared__ alignas(16) u16 Bs[2][128 * LDSTRIDE];

  int z  = blockIdx.z;
  int bb = z / nH, hh = z - bb * nH;
  A  += (long)bb * aOffB + (long)hh * aOffH;
  BT += (long)bb * bOffB + (long)hh * bOffH;
  long cOff = (long)bb * cOffB + (long)hh * cOffH;

  const int t    = threadIdx.x;
  const int lane = t & 31;
  const int wave = t >> 5;
  const int wm   = wave >> 1;        // 0..3
  const int wn   = wave & 1;         // 0..1
  const int BN   = 32 * WN;
  const int rowBlk = blockIdx.y * 128;
  const int colBlk = blockIdx.x * BN;

  const u16* Ag = A  + (long)rowBlk * lda;
  const u16* Bg = BT + (long)colBlk * ldb;

  v8f acc[2][WN];
  #pragma unroll
  for (int i = 0; i < 2; ++i)
    #pragma unroll
    for (int j = 0; j < WN; ++j) acc[i][j] = (v8f){};

  // prologue: stage K-slab 0 into buffer 0
  copy_tile<128>(Ag, lda, As[0], t);
  copy_tile<32 * WN>(Bg, ldb, Bs[0], t);

  const int KT = K >> 5;
  for (int kt = 0; kt < KT; ++kt) {
    int cur = kt & 1;
#if HAVE_ASYNC_LDS
    ASYNC_WAIT0();              // my copies for buf[cur] have landed in LDS
#endif
    __syncthreads();            // everyone's copies visible

    if (kt + 1 < KT) {          // overlap: stage next slab while computing
      copy_tile<128>(Ag + (kt + 1) * 32, lda, As[cur ^ 1], t);
      copy_tile<32 * WN>(Bg + (kt + 1) * 32, ldb, Bs[cur ^ 1], t);
    }

    v16bf af0 = lds_frag(&As[cur][(wm * 32) * LDSTRIDE], lane);
    v16bf af1 = lds_frag(&As[cur][(wm * 32 + 16) * LDSTRIDE], lane);
    #pragma unroll
    for (int j = 0; j < WN; ++j) {
      v16bf bfj = lds_frag(&Bs[cur][(wn * WN * 16 + j * 16) * LDSTRIDE], lane);
      acc[0][j] = __builtin_amdgcn_wmma_f32_16x16x32_bf16(false, af0, false, bfj, (short)0, acc[0][j], false, false);
      acc[1][j] = __builtin_amdgcn_wmma_f32_16x16x32_bf16(false, af1, false, bfj, (short)0, acc[1][j], false, false);
    }
  }

  // D layout: lane -> n = lane&15, row = 8*(lane>>4) + vgpr index r
  int n    = lane & 15;
  int half = lane >> 4;
  #pragma unroll
  for (int j = 0; j < WN; ++j) {
    int col = colBlk + wn * WN * 16 + j * 16 + n;
    float bvj = bias ? bias[col] : 0.0f;
    #pragma unroll
    for (int r = 0; r < 8; ++r) {
      int row0 = rowBlk + wm * 32 + half * 8 + r;
      long o0 = cOff + (long)row0 * ldc + col;
      long o1 = o0 + (long)16 * ldc;
      float v0 = acc[0][j][r] + bvj;
      float v1 = acc[1][j][r] + bvj;
      if (Cf) { Cf[o0] = v0; Cf[o1] = v1; }
      if (Cb) { Cb[o0] = f2bf(v0); Cb[o1] = f2bf(v1); }
    }
  }
}

// f32 -> bf16 convert (+row mask); optional seq-slice gather
__global__ __launch_bounds__(256) void cvt_bf16(
    const float* __restrict__ in, const float* __restrict__ mask, u16* __restrict__ out,
    int n, int rowLen, int sliceLen, int sliceStart, int seq)
{
  int i = blockIdx.x * blockDim.x + threadIdx.x;
  if (i >= n) return;
  int row = i / rowLen;
  int col = i - row * rowLen;
  int srcRow = row;
  if (sliceLen > 0) {
    int b = row / sliceLen;
    int l = row - b * sliceLen;
    srcRow = b * seq + sliceStart + l;
  }
  float v = in[(long)srcRow * rowLen + col];
  if (mask) v *= mask[srcRow];
  out[i] = f2bf(v);
}

// DxD f32 row-major -> bf16 transposed via LDS tile
__global__ __launch_bounds__(256) void transpose_cvt(
    const float* __restrict__ in, u16* __restrict__ out, int D)
{
  __shared__ float tile[32][33];
  int tx = threadIdx.x & 31;
  int ty = threadIdx.x >> 5;           // 0..7
  int bx = blockIdx.x * 32;
  int by = blockIdx.y * 32;
  #pragma unroll
  for (int i = 0; i < 32; i += 8)
    tile[ty + i][tx] = in[(long)(by + ty + i) * D + bx + tx];
  __syncthreads();
  #pragma unroll
  for (int i = 0; i < 32; i += 8)
    out[(long)(bx + ty + i) * D + by + tx] = f2bf(tile[tx][ty + i]);
}

// fvT[b][h][d][l] = Vp[(b*128+l)*1024 + h*64 + d]
__global__ __launch_bounds__(256) void build_fvT(
    const u16* __restrict__ Vp, u16* __restrict__ fvT, int n)
{
  int i = blockIdx.x * blockDim.x + threadIdx.x;
  if (i >= n) return;
  int l = i & (LLEN - 1);
  int d = (i >> 7) & (DEPTH - 1);
  int z = i >> 13;                 // b*16 + h
  int b = z >> 4, h = z & 15;
  fvT[i] = Vp[(long)(b * LLEN + l) * D_MODEL + h * DEPTH + d];
}

// In-place scaled+masked softmax over rows of 128 (wave32 per row) + bf16 copy
__global__ __launch_bounds__(256) void softmax_rows(
    float* __restrict__ logits, u16* __restrict__ attnb, int nrows)
{
  int lane = threadIdx.x & 31;
  int wave = threadIdx.x >> 5;
  int row = blockIdx.x * 8 + wave;
  if (row >= nrows) return;
  int q = row & (SEQ - 1);
  long base = (long)row * LLEN;
  float4 v = *(const float4*)(logits + base + lane * 4);
  const float scale = 0.125f;         // 1/sqrt(64)
  int c = lane * 4;
  float x0 = (c + 0 > q) ? -1.0e9f : v.x * scale;
  float x1 = (c + 1 > q) ? -1.0e9f : v.y * scale;
  float x2 = (c + 2 > q) ? -1.0e9f : v.z * scale;
  float x3 = (c + 3 > q) ? -1.0e9f : v.w * scale;
  float m = fmaxf(fmaxf(x0, x1), fmaxf(x2, x3));
  for (int off = 16; off >= 1; off >>= 1) m = fmaxf(m, __shfl_xor(m, off, 32));
  x0 = __expf(x0 - m); x1 = __expf(x1 - m);
  x2 = __expf(x2 - m); x3 = __expf(x3 - m);
  float s = x0 + x1 + x2 + x3;
  for (int off = 16; off >= 1; off >>= 1) s += __shfl_xor(s, off, 32);
  float inv = 1.0f / s;
  x0 *= inv; x1 *= inv; x2 *= inv; x3 *= inv;
  float4 o; o.x = x0; o.y = x1; o.z = x2; o.w = x3;
  *(float4*)(logits + base + lane * 4) = o;
  uint2 pp;
  pp.x = (u32)f2bf(x0) | ((u32)f2bf(x1) << 16);
  pp.y = (u32)f2bf(x2) | ((u32)f2bf(x3) << 16);
  *(uint2*)(attnb + base + lane * 4) = pp;
}

extern "C" void kernel_launch(void* const* d_in, const int* in_sizes, int n_in,
                              void* d_out, int out_size, void* d_ws, size_t ws_size,
                              hipStream_t stream) {
  (void)in_sizes; (void)n_in; (void)out_size; (void)ws_size;
  const float* q  = (const float*)d_in[0];
  const float* k  = (const float*)d_in[1];
  const float* v  = (const float*)d_in[2];
  const float* qmask = (const float*)d_in[3];
  const float* kmask = (const float*)d_in[4];
  const float* vmask = (const float*)d_in[5];
  const float* wq = (const float*)d_in[6];
  const float* bq = (const float*)d_in[7];
  const float* wk = (const float*)d_in[8];
  const float* bk = (const float*)d_in[9];
  const float* wv = (const float*)d_in[10];
  const float* bv = (const float*)d_in[11];
  const float* wo = (const float*)d_in[12];
  const float* bo = (const float*)d_in[13];

  float* out  = (float*)d_out;                                 // (4,2048,1024)
  float* attn = out + (size_t)BATCH * SEQ * D_MODEL;           // (4,16,2048,128)

  char* ws = (char*)d_ws;
  size_t off = 0;
  auto alloc = [&](size_t bytes) -> char* {
    char* p = ws + off;
    off = (off + bytes + 255) & ~(size_t)255;
    return p;
  };
  const size_t MQ = (size_t)BATCH * SEQ;    // 8192
  const size_t MK = (size_t)BATCH * LLEN;   // 512

  u16* qbf = (u16*)alloc(MQ * D_MODEL * 2);
  u16* wqT = (u16*)alloc((size_t)D_MODEL * D_MODEL * 2);
  u16* wkT = (u16*)alloc((size_t)D_MODEL * D_MODEL * 2);
  u16* wvT = (u16*)alloc((size_t)D_MODEL * D_MODEL * 2);
  u16* woT = (u16*)alloc((size_t)D_MODEL * D_MODEL * 2);
  u16* ksl = (u16*)alloc(MK * D_MODEL * 2);
  u16* vsl = (u16*)alloc(MK * D_MODEL * 2);
  u16* Qp  = (u16*)alloc(MQ * D_MODEL * 2);
  u16* Kp  = (u16*)alloc(MK * D_MODEL * 2);
  u16* Vp  = (u16*)alloc(MK * D_MODEL * 2);
  u16* fvT = (u16*)alloc((size_t)BATCH * NUM_HEADS * DEPTH * LLEN * 2);
  u16* attnb = (u16*)alloc((size_t)BATCH * NUM_HEADS * SEQ * LLEN * 2);
  u16* ctx = (u16*)alloc(MQ * D_MODEL * 2);

  // 1) converts
  int nQ = (int)(MQ * D_MODEL);
  cvt_bf16<<<(nQ + 255) / 256, 256, 0, stream>>>(q, qmask, qbf, nQ, D_MODEL, 0, 0, SEQ);
  int nS = (int)(MK * D_MODEL);
  cvt_bf16<<<(nS + 255) / 256, 256, 0, stream>>>(k, kmask, ksl, nS, D_MODEL, LLEN, LSTART, SEQ);
  cvt_bf16<<<(nS + 255) / 256, 256, 0, stream>>>(v, vmask, vsl, nS, D_MODEL, LLEN, LSTART, SEQ);
  dim3 tg(D_MODEL / 32, D_MODEL / 32);
  transpose_cvt<<<tg, 256, 0, stream>>>(wq, wqT, D_MODEL);
  transpose_cvt<<<tg, 256, 0, stream>>>(wk, wkT, D_MODEL);
  transpose_cvt<<<tg, 256, 0, stream>>>(wv, wvT, D_MODEL);
  transpose_cvt<<<tg, 256, 0, stream>>>(wo, woT, D_MODEL);

  // 2) projections (only the used 512-row K/V slice gets projected)
  wmma_gemm_lds<4><<<dim3(D_MODEL / 128, (unsigned)(MQ / 128), 1), 256, 0, stream>>>(
      qbf, wqT, nullptr, Qp, bq, D_MODEL, D_MODEL, D_MODEL, D_MODEL,
      0, 0, 0, 0, 0, 0, 1);
  wmma_gemm_lds<4><<<dim3(D_MODEL / 128, (unsigned)(MK / 128), 1), 256, 0, stream>>>(
      ksl, wkT, nullptr, Kp, bk, D_MODEL, D_MODEL, D_MODEL, D_MODEL,
      0, 0, 0, 0, 0, 0, 1);
  wmma_gemm_lds<4><<<dim3(D_MODEL / 128, (unsigned)(MK / 128), 1), 256, 0, stream>>>(
      vsl, wvT, nullptr, Vp, bv, D_MODEL, D_MODEL, D_MODEL, D_MODEL,
      0, 0, 0, 0, 0, 0, 1);

  int nF = BATCH * NUM_HEADS * DEPTH * LLEN;
  build_fvT<<<(nF + 255) / 256, 256, 0, stream>>>(Vp, fvT, nF);

  // 3) logits = qh @ fk^T  (per (b,h): 2048x128, K=64) -> f32 into d_out attn region
  wmma_gemm_lds<4><<<dim3(LLEN / 128, SEQ / 128, BATCH * NUM_HEADS), 256, 0, stream>>>(
      Qp, Kp, attn, nullptr, nullptr, DEPTH, D_MODEL, D_MODEL, LLEN,
      (long)SEQ * D_MODEL, (long)DEPTH,
      (long)LLEN * D_MODEL, (long)DEPTH,
      (long)NUM_HEADS * SEQ * LLEN, (long)SEQ * LLEN,
      NUM_HEADS);

  // 4) softmax (scale + causal mask over first 127 rows), in place + bf16 copy
  int nrows = BATCH * NUM_HEADS * SEQ;
  softmax_rows<<<nrows / 8, 256, 0, stream>>>(attn, attnb, nrows);

  // 5) context = attn @ fv  (per (b,h): 2048x64, K=128) -> bf16 (b,s,1024) layout
  wmma_gemm_lds<2><<<dim3(DEPTH / 64, SEQ / 128, BATCH * NUM_HEADS), 256, 0, stream>>>(
      attnb, fvT, nullptr, ctx, nullptr, LLEN, LLEN, LLEN, D_MODEL,
      (long)NUM_HEADS * SEQ * LLEN, (long)SEQ * LLEN,
      (long)NUM_HEADS * DEPTH * LLEN, (long)DEPTH * LLEN,
      (long)SEQ * D_MODEL, (long)DEPTH,
      NUM_HEADS);

  // 6) output = ctx @ wo + bo -> f32 d_out
  wmma_gemm_lds<4><<<dim3(D_MODEL / 128, (unsigned)(MQ / 128), 1), 256, 0, stream>>>(
      ctx, woT, out, nullptr, bo, D_MODEL, D_MODEL, D_MODEL, D_MODEL,
      0, 0, 0, 0, 0, 0, 1);
}
